// Attention_6786048328639
// MI455X (gfx1250) — matrix operand order repaired and checked by reading, not compiled, not run on hardware
//
#include <hip/hip_runtime.h>

#define B_DIM 64
#define S_DIM 1024
#define H_DIM 512
#define NEGV (-1e10f)

typedef __attribute__((ext_vector_type(16))) __bf16 v16bf;
typedef __attribute__((ext_vector_type(8)))  float  v8f;

// round-to-nearest-even fp32 -> bf16, packed pair into one dword
static __device__ __forceinline__ unsigned short f2bf(float f) {
    unsigned u = __float_as_uint(f);
    unsigned r = u + 0x7FFFu + ((u >> 16) & 1u);
    return (unsigned short)(r >> 16);
}
static __device__ __forceinline__ unsigned pack2(float a, float b) {
    return (unsigned)f2bf(a) | ((unsigned)f2bf(b) << 16);
}

// ---------------------------------------------------------------------------
// Prep 1: W2[h][k] = bf16(attn_w[h][H+k]); 512x512 bf16 (512 KB, L2-resident)
// ---------------------------------------------------------------------------
__global__ void prep_w2(const float* __restrict__ attn_w, unsigned* __restrict__ w2) {
    int i  = blockIdx.x * blockDim.x + threadIdx.x;   // 0..131071, 2 elems each
    int h  = i >> 8;                                  // 256 packed dwords / row
    int k2 = (i & 255) * 2;
    const float* src = attn_w + (size_t)h * (2 * H_DIM) + H_DIM + k2;
    w2[i] = pack2(src[0], src[1]);
}

// ---------------------------------------------------------------------------
// Prep 2: base_vw[b][h] = { hidden[b]·attn_w[h, :H] + attn_b[h], v_w[h] }
// Tiny (33 MFLOP); scalar fp32, weights/hidden hit L2.
// ---------------------------------------------------------------------------
__global__ void prep_base(const float* __restrict__ hidden,
                          const float* __restrict__ attn_w,
                          const float* __restrict__ attn_b,
                          const float* __restrict__ v_w,
                          float2* __restrict__ base_vw) {
    int i = blockIdx.x * blockDim.x + threadIdx.x;    // 0..32767
    int b = i >> 9;
    int h = i & 511;
    const float* hid = hidden + (size_t)b * H_DIM;
    const float* wr  = attn_w + (size_t)h * (2 * H_DIM);
    float s = attn_b[h];
#pragma unroll 8
    for (int k = 0; k < H_DIM; ++k) s += hid[k] * wr[k];
    base_vw[i] = make_float2(s, v_w[h]);
}

// ---------------------------------------------------------------------------
// Main: per wave a 16x512 A-tile (fixed b, 16 consecutive s).
// BF16 WMMA over 32 N-tiles x 16 K-tiles, fused tanh + v-dot epilogue,
// ds_swizzle half-wave reduction, masked logits to d_out.
// ---------------------------------------------------------------------------
__global__ void __launch_bounds__(256)
attn_gemm(const float* __restrict__ enc,              // [S,B,H] fp32
          const unsigned short* __restrict__ w2,      // [H][H] bf16 row-major
          const float2* __restrict__ base_vw,         // [B][H]
          const int* __restrict__ attn_mask,          // [B][S]
          float* __restrict__ logits)                 // [B][S] (d_out)
{
    const int wave = threadIdx.x >> 5;
    const int lane = threadIdx.x & 31;
    const int ln   = lane & 15;
    const int hi   = lane >> 4;
    const int tile = blockIdx.x * 8 + wave;           // 0..4095
    const int b    = tile >> 6;
    const int s0   = (tile & 63) << 4;

    // --- Stage A in registers: lane owns row m=ln (s = s0+ln). Per K-tile the
    // ISA layout wants two 16B chunks: bytes {hi*16, 32+hi*16} of the 64B tile
    // (fp32 source: floats {kt*32+hi*8, kt*32+16+hi*8}). Each lane's 4 b128
    // loads per kt read two 32B runs; lanes ln and ln+16 together cover the kt's 128B.
    const float* arow = enc + ((size_t)(s0 + ln) * B_DIM + b) * H_DIM + hi * 8;
    v16bf afrag[16];
#pragma unroll
    for (int kt = 0; kt < 16; ++kt) {
        const float4* p0 = (const float4*)(arow + kt * 32);
        const float4* p1 = (const float4*)(arow + kt * 32 + 16);
        float4 a0 = p0[0], a1 = p0[1], c0 = p1[0], c1 = p1[1];
        union { v16bf v; unsigned u[8]; } ua;
        ua.u[0] = pack2(a0.x, a0.y); ua.u[1] = pack2(a0.z, a0.w);
        ua.u[2] = pack2(a1.x, a1.y); ua.u[3] = pack2(a1.z, a1.w);
        ua.u[4] = pack2(c0.x, c0.y); ua.u[5] = pack2(c0.z, c0.w);
        ua.u[6] = pack2(c1.x, c1.y); ua.u[7] = pack2(c1.z, c1.w);
        afrag[kt] = ua.v;
    }

    float part[8];
#pragma unroll
    for (int j = 0; j < 8; ++j) part[j] = 0.f;

    for (int nt = 0; nt < 32; ++nt) {
        // B fragment: lane = column n = nt*16+ln; per K-tile two 16B chunks
        // at byte offsets {hi*16, 32+hi*16} — W2 row-major means no transpose.
        const unsigned short* brow = w2 + (size_t)(nt * 16 + ln) * H_DIM + hi * 8;
        v8f acc = {};
#pragma unroll
        for (int kt = 0; kt < 16; ++kt) {
            const uint4* bp = (const uint4*)(brow + kt * 32);
            union { v16bf v; uint4 q[2]; } ub;
            ub.q[0] = bp[0];
            ub.q[1] = bp[2];
            acc = __builtin_amdgcn_wmma_f32_16x16x32_bf16(
                false, afrag[kt], false, ub.v, (short)0, acc, false, false);
        }
        // epilogue: C elem j of this lane is (m = j+8*hi, n = nt*16+ln)
        float2 bv = base_vw[b * H_DIM + nt * 16 + ln];
#pragma unroll
        for (int j = 0; j < 8; ++j)
            part[j] += tanhf(acc[j] + bv.x) * bv.y;
    }

    // reduce over n across each 16-lane half (xor 1,2,4,8 stays in-half)
#pragma unroll
    for (int j = 0; j < 8; ++j) {
        float v = part[j];
        v += __int_as_float(__builtin_amdgcn_ds_swizzle(__float_as_int(v), 0x041F));
        v += __int_as_float(__builtin_amdgcn_ds_swizzle(__float_as_int(v), 0x081F));
        v += __int_as_float(__builtin_amdgcn_ds_swizzle(__float_as_int(v), 0x101F));
        v += __int_as_float(__builtin_amdgcn_ds_swizzle(__float_as_int(v), 0x201F));
        part[j] = v;
    }
    if (ln == 0) {  // lanes 0 and 16 hold rows m=j and m=j+8
#pragma unroll
        for (int j = 0; j < 8; ++j) {
            int s = s0 + hi * 8 + j;
            float val = part[j];
            if (attn_mask[b * S_DIM + s] == 0) val = NEGV;
            logits[(size_t)b * S_DIM + s] = val;
        }
    }
}

// ---------------------------------------------------------------------------
// In-place row softmax over S=1024 (one block per b). All-masked rows give
// max=-1e10 -> exp(0)=1 -> uniform, matching JAX.
// ---------------------------------------------------------------------------
__global__ void __launch_bounds__(256) softmax_rows(float* __restrict__ out) {
    __shared__ float red[8];
    const int b = blockIdx.x;
    const int t = threadIdx.x;
    float x[4];
#pragma unroll
    for (int j = 0; j < 4; ++j) x[j] = out[(size_t)b * S_DIM + t + 256 * j];

    float m = fmaxf(fmaxf(x[0], x[1]), fmaxf(x[2], x[3]));
    m = fmaxf(m, __int_as_float(__builtin_amdgcn_ds_swizzle(__float_as_int(m), 0x041F)));
    m = fmaxf(m, __int_as_float(__builtin_amdgcn_ds_swizzle(__float_as_int(m), 0x081F)));
    m = fmaxf(m, __int_as_float(__builtin_amdgcn_ds_swizzle(__float_as_int(m), 0x101F)));
    m = fmaxf(m, __int_as_float(__builtin_amdgcn_ds_swizzle(__float_as_int(m), 0x201F)));
    m = fmaxf(m, __int_as_float(__builtin_amdgcn_ds_swizzle(__float_as_int(m), 0x401F)));
    if ((t & 31) == 0) red[t >> 5] = m;
    __syncthreads();
    float bm = red[0];
#pragma unroll
    for (int i = 1; i < 8; ++i) bm = fmaxf(bm, red[i]);
    __syncthreads();

    float s = 0.f;
#pragma unroll
    for (int j = 0; j < 4; ++j) { x[j] = __expf(x[j] - bm); s += x[j]; }
    s += __int_as_float(__builtin_amdgcn_ds_swizzle(__float_as_int(s), 0x041F));
    s += __int_as_float(__builtin_amdgcn_ds_swizzle(__float_as_int(s), 0x081F));
    s += __int_as_float(__builtin_amdgcn_ds_swizzle(__float_as_int(s), 0x101F));
    s += __int_as_float(__builtin_amdgcn_ds_swizzle(__float_as_int(s), 0x201F));
    s += __int_as_float(__builtin_amdgcn_ds_swizzle(__float_as_int(s), 0x401F));
    if ((t & 31) == 0) red[t >> 5] = s;
    __syncthreads();
    float tot = 0.f;
#pragma unroll
    for (int i = 0; i < 8; ++i) tot += red[i];
    float inv = 1.0f / tot;
#pragma unroll
    for (int j = 0; j < 4; ++j) out[(size_t)b * S_DIM + t + 256 * j] = x[j] * inv;
}

// ---------------------------------------------------------------------------
extern "C" void kernel_launch(void* const* d_in, const int* in_sizes, int n_in,
                              void* d_out, int out_size, void* d_ws, size_t ws_size,
                              hipStream_t stream) {
    // setup_inputs order:
    // 0:t 1:hidden 2:encoder_outputs 3:attn_mask 4:src_gps_seqs 5:src
    // 6:src_rids 7:input_id 8:trg_gps_seqs 9:attn_w 10:attn_b 11:v_w
    const float* hidden   = (const float*)d_in[1];
    const float* enc      = (const float*)d_in[2];
    const int*   attnmask = (const int*)d_in[3];
    const float* attn_w   = (const float*)d_in[9];
    const float* attn_b   = (const float*)d_in[10];
    const float* v_w      = (const float*)d_in[11];

    unsigned* w2      = (unsigned*)d_ws;                          // 512 KB bf16 W2
    float2*   base_vw = (float2*)((char*)d_ws + 512 * 1024);      // 256 KB
    float*    out     = (float*)d_out;

    prep_w2  <<<512, 256, 0, stream>>>(attn_w, w2);
    prep_base<<<128, 256, 0, stream>>>(hidden, attn_w, attn_b, v_w, base_vw);
    attn_gemm<<<512, 256, 0, stream>>>(enc, (const unsigned short*)w2,
                                       base_vw, attnmask, out);
    softmax_rows<<<B_DIM, 256, 0, stream>>>(out);
}
